// Decoder_66254165508798
// MI455X (gfx1250) — matrix operand order repaired and checked by reading, not compiled, not run on hardware
//
#include <hip/hip_runtime.h>
#include <hip/hip_bf16.h>
#include <math.h>

// ---------------------------------------------------------------------------
// Types for CDNA5 WMMA (wave32, 16x16x32 bf16 -> f32 accum)
// ---------------------------------------------------------------------------
typedef __attribute__((ext_vector_type(16))) __bf16 v16bf;
typedef __attribute__((ext_vector_type(8)))  __bf16 v8bf;
typedef __attribute__((ext_vector_type(8)))  float  v8f;

#define B_SZ      64
#define T_ENC     512
#define D_ENC     512
#define T_DEC     500
#define NUM_ITER  250
#define PRE_DIM   256
#define Q_DIM     1024
#define DD_DIM    1024
#define ATTN_DIM  128
#define K1        1792   /* 256 pre + 512 ctx + 1024 h1 */
#define K2        2560   /* 1024 h1 + 512 ctx + 1024 h2 */
#define N_GATE    4096
#define FRAME     80
#define R_FR      160    /* R*FRAME */

__device__ __forceinline__ float sigmoidf(float x) { return 1.0f / (1.0f + expf(-x)); }

// ---------------------------------------------------------------------------
// Zero-init helpers (workspace state must be deterministic per launch)
// ---------------------------------------------------------------------------
__global__ void zero_f32_k(float* p, int n) {
  int i = blockIdx.x * 256 + threadIdx.x;
  if (i < n) p[i] = 0.0f;
}
__global__ void zero_bf16_k(__bf16* p, int n) {
  int i = blockIdx.x * 256 + threadIdx.x;
  if (i < n) p[i] = (__bf16)0.0f;
}

// ---------------------------------------------------------------------------
// Pack fp32 weights [K x N] (stacked Wx;Wh) into bf16 WMMA B-fragment layout:
//   packed[ ((nt*nK + kt)*32 + lane)*16 + half ]
//   n = nt*16 + (lane&15),  k = kt*32 + (lane>=16 ? 8 : 0) + (half<8 ? half : 8+half)
// so each lane of a wave loads its whole fragment with ONE 32B load.
// ---------------------------------------------------------------------------
__global__ void pack_weights_k(const float* __restrict__ wx,
                               const float* __restrict__ wh,
                               int kx, int kh, int N, __bf16* __restrict__ out) {
  long long idx = (long long)blockIdx.x * 256 + threadIdx.x;
  long long K = kx + kh;
  if (idx >= K * N) return;
  int half = (int)(idx & 15);
  int lane = (int)((idx >> 4) & 31);
  long long tile = idx >> 9;
  int nK = (int)(K >> 5);
  int kt = (int)(tile % nK);
  int nt = (int)(tile / nK);
  int n = nt * 16 + (lane & 15);
  int k = kt * 32 + ((lane >> 4) << 3) + (half < 8 ? half : 8 + half);
  float v = (k < kx) ? wx[(long long)k * N + n] : wh[(long long)(k - kx) * N + n];
  out[idx] = (__bf16)v;
}

// ---------------------------------------------------------------------------
// WMMA GEMM:  C[64,N] = A[64,K](bf16 row-major) @ Wpacked + bias
// M-register-blocked: one wave owns ONE N-tile and ALL FOUR M-tiles (M=64),
// so each B (weight) fragment is loaded once per 4 WMMAs -> minimum L2
// weight traffic (the binding resource for these skinny GEMMs), and each
// K-tile exposes 4 independent WMMAs to hide load latency.
// 8 waves (8 N-tiles) per 256-thread block. K multiple of 32.
// ---------------------------------------------------------------------------
__global__ __launch_bounds__(256)
void gemm_wmma_bf16_k(const __bf16* __restrict__ A,
                      const __bf16* __restrict__ Wp,
                      const float* __restrict__ bias,
                      float* __restrict__ C, int K, int N) {
  int lane = threadIdx.x & 31;
  int wave = threadIdx.x >> 5;
  int nt = blockIdx.x * 8 + wave;
  if (nt >= (N >> 4)) return;
  int nK = K >> 5;

  int kbase = (lane >> 4) << 3;              // 0 or 8 (A-fragment layout)
  const __bf16* Arow = A + (long long)(lane & 15) * K + kbase;   // M-tile 0
  const __bf16* Bp   = Wp + ((long long)nt * nK) * 512 + (long long)lane * 16;
  const long long ms = (long long)16 * K;    // M-tile row stride in A

  v8f acc0 = {}, acc1 = {}, acc2 = {}, acc3 = {};
  for (int kt = 0; kt < nK; ++kt) {
    __builtin_prefetch(Bp + 512, 0, 1);      // next K-tile of B -> global_prefetch
    v16bf b = *(const v16bf*)(Bp);

    v8bf a0l = *(const v8bf*)(Arow);
    v8bf a0h = *(const v8bf*)(Arow + 16);
    v8bf a1l = *(const v8bf*)(Arow + ms);
    v8bf a1h = *(const v8bf*)(Arow + ms + 16);
    v8bf a2l = *(const v8bf*)(Arow + 2 * ms);
    v8bf a2h = *(const v8bf*)(Arow + 2 * ms + 16);
    v8bf a3l = *(const v8bf*)(Arow + 3 * ms);
    v8bf a3h = *(const v8bf*)(Arow + 3 * ms + 16);

    v16bf a0 = __builtin_shufflevector(a0l, a0h, 0,1,2,3,4,5,6,7,8,9,10,11,12,13,14,15);
    v16bf a1 = __builtin_shufflevector(a1l, a1h, 0,1,2,3,4,5,6,7,8,9,10,11,12,13,14,15);
    v16bf a2 = __builtin_shufflevector(a2l, a2h, 0,1,2,3,4,5,6,7,8,9,10,11,12,13,14,15);
    v16bf a3 = __builtin_shufflevector(a3l, a3h, 0,1,2,3,4,5,6,7,8,9,10,11,12,13,14,15);

    acc0 = __builtin_amdgcn_wmma_f32_16x16x32_bf16(false, a0, false, b, (short)0, acc0, false, false);
    acc1 = __builtin_amdgcn_wmma_f32_16x16x32_bf16(false, a1, false, b, (short)0, acc1, false, false);
    acc2 = __builtin_amdgcn_wmma_f32_16x16x32_bf16(false, a2, false, b, (short)0, acc2, false, false);
    acc3 = __builtin_amdgcn_wmma_f32_16x16x32_bf16(false, a3, false, b, (short)0, acc3, false, false);

    Arow += 32;
    Bp   += 512;
  }

  int col   = nt * 16 + (lane & 15);
  int rsub  = (lane >> 4) << 3;              // lanes 16..31 hold rows M=8..15
  float bv = bias ? bias[col] : 0.0f;
  float* Cc = C + col;
#pragma unroll
  for (int r = 0; r < 8; ++r) Cc[(long long)(rsub + r) * N]      = acc0[r] + bv;
#pragma unroll
  for (int r = 0; r < 8; ++r) Cc[(long long)(16 + rsub + r) * N] = acc1[r] + bv;
#pragma unroll
  for (int r = 0; r < 8; ++r) Cc[(long long)(32 + rsub + r) * N] = acc2[r] + bv;
#pragma unroll
  for (int r = 0; r < 8; ++r) Cc[(long long)(48 + rsub + r) * N] = acc3[r] + bv;
}

// ---------------------------------------------------------------------------
// Prenet (one-time): layer1 with implicit zero frame prepend, then layer2->bf16
// ---------------------------------------------------------------------------
__global__ void prenet1_k(const float* __restrict__ frames,
                          const float* __restrict__ w1,
                          float* __restrict__ tmp) {
  int idx = blockIdx.x * 256 + threadIdx.x;       // B*NUM_ITER*256
  if (idx >= B_SZ * NUM_ITER * PRE_DIM) return;
  int j = idx & 255;
  int t = (idx >> 8) % NUM_ITER;
  int b = idx / (NUM_ITER * PRE_DIM);
  float s = 0.0f;
  if (t > 0) {
    const float* x = frames + ((long long)b * T_DEC + (t - 1)) * FRAME;
    for (int k = 0; k < FRAME; ++k) s += x[k] * w1[k * PRE_DIM + j];
  }
  tmp[idx] = fmaxf(s, 0.0f);
}

__global__ void prenet2_k(const float* __restrict__ tmp,
                          const float* __restrict__ w2,
                          __bf16* __restrict__ pre_bf) {
  int idx = blockIdx.x * 256 + threadIdx.x;
  if (idx >= B_SZ * NUM_ITER * PRE_DIM) return;
  int j  = idx & 255;
  int bt = idx >> 8;
  const float* x = tmp + (long long)bt * PRE_DIM;
  float s = 0.0f;
  for (int k = 0; k < PRE_DIM; ++k) s += x[k] * w2[k * PRE_DIM + j];
  pre_bf[idx] = (__bf16)fmaxf(s, 0.0f);
}

// processed_values = memory @ attn_wi  (one-time)
__global__ void pv_k(const float* __restrict__ memory,
                     const float* __restrict__ wi,
                     float* __restrict__ pv) {
  int idx = blockIdx.x * 256 + threadIdx.x;       // B*T_ENC*ATTN_DIM
  if (idx >= B_SZ * T_ENC * ATTN_DIM) return;
  int d  = idx & 127;
  int bt = idx >> 7;
  const float* m = memory + (long long)bt * D_ENC;
  float s = 0.0f;
  for (int k = 0; k < D_ENC; ++k) s += m[k] * wi[k * ATTN_DIM + d];
  pv[idx] = s;
}

// ---------------------------------------------------------------------------
// Per-step activation concat -> bf16  (A for the stacked-W GEMMs)
// ---------------------------------------------------------------------------
__global__ void assembleA1_k(const __bf16* __restrict__ pre_bf,
                             const float* __restrict__ context,
                             const __bf16* __restrict__ h1_bf,
                             __bf16* __restrict__ A1, int t) {
  int idx = blockIdx.x * 256 + threadIdx.x;       // B*K1
  if (idx >= B_SZ * K1) return;
  int j = idx % K1, b = idx / K1;
  __bf16 v;
  if (j < PRE_DIM)             v = pre_bf[((long long)b * NUM_ITER + t) * PRE_DIM + j];
  else if (j < PRE_DIM + D_ENC) v = (__bf16)context[b * D_ENC + (j - PRE_DIM)];
  else                         v = h1_bf[b * Q_DIM + (j - PRE_DIM - D_ENC)];
  A1[idx] = v;
}

__global__ void assembleA2_k(const __bf16* __restrict__ h1_bf,
                             const float* __restrict__ context,
                             const __bf16* __restrict__ h2_bf,
                             __bf16* __restrict__ A2) {
  int idx = blockIdx.x * 256 + threadIdx.x;       // B*K2
  if (idx >= B_SZ * K2) return;
  int j = idx % K2, b = idx / K2;
  __bf16 v;
  if (j < Q_DIM)               v = h1_bf[b * Q_DIM + j];
  else if (j < Q_DIM + D_ENC)  v = (__bf16)context[b * D_ENC + (j - Q_DIM)];
  else                         v = h2_bf[b * DD_DIM + (j - Q_DIM - D_ENC)];
  A2[idx] = v;
}

// ---------------------------------------------------------------------------
// LSTM gate nonlinearity (Keras order i,f,g,o), updates h,c and writes bf16 h
// ---------------------------------------------------------------------------
__global__ void lstm_update_k(const float* __restrict__ z,
                              float* __restrict__ h, float* __restrict__ c,
                              __bf16* __restrict__ h_bf) {
  int idx = blockIdx.x * 256 + threadIdx.x;       // B*1024
  if (idx >= B_SZ * Q_DIM) return;
  int b = idx / Q_DIM, n = idx % Q_DIM;
  const float* zr = z + (long long)b * N_GATE;
  float ig = sigmoidf(zr[n]);
  float fg = sigmoidf(zr[n + Q_DIM]);
  float gg = tanhf(zr[n + 2 * Q_DIM]);
  float og = sigmoidf(zr[n + 3 * Q_DIM]);
  float cn = fg * c[idx] + ig * gg;
  float hn = og * tanhf(cn);
  c[idx] = cn;
  h[idx] = hn;
  h_bf[idx] = (__bf16)hn;
}

// ---------------------------------------------------------------------------
// Location-sensitive attention + softmax + context, one block per batch row.
// Also writes the attention row into d_out and updates a_cum/a_old.
// ---------------------------------------------------------------------------
__global__ __launch_bounds__(256)
void attention_k(const float* __restrict__ proc_q,   // [B,128]
                 const float* __restrict__ pv,       // [B,512,128]
                 const float* __restrict__ conv_w,   // [31,2,32]
                 const float* __restrict__ dense_w,  // [32,128]
                 const float* __restrict__ v_w,      // [128]
                 const float* __restrict__ v_b,      // [1]
                 const float* __restrict__ memory,   // [B,512,512]
                 float* __restrict__ a_cum, float* __restrict__ a_old,
                 float* __restrict__ context,        // [B,512]
                 float* __restrict__ attn_out, int t) {
  __shared__ float q[ATTN_DIM], vw[ATTN_DIM];
  __shared__ float cw[31 * 2 * 32];
  __shared__ float dw[32 * ATTN_DIM];
  __shared__ float sc[T_ENC];
  __shared__ float red[256];
  int b = blockIdx.x, tid = threadIdx.x;

  for (int i = tid; i < ATTN_DIM; i += 256) { q[i] = proc_q[b * ATTN_DIM + i]; vw[i] = v_w[i]; }
  for (int i = tid; i < 31 * 64; i += 256) cw[i] = conv_w[i];
  for (int i = tid; i < 32 * ATTN_DIM; i += 256) dw[i] = dense_w[i];
  __syncthreads();

  const float* ao = a_old + (long long)b * T_ENC;
  const float* ac = a_cum + (long long)b * T_ENC;
  float my_s[2];
#pragma unroll
  for (int half = 0; half < 2; ++half) {
    int ti = tid + half * 256;
    float lf[32];
#pragma unroll
    for (int f = 0; f < 32; ++f) lf[f] = 0.0f;
    for (int tap = 0; tap < 31; ++tap) {
      int src = ti + tap - 15;
      if (src >= 0 && src < T_ENC) {
        float va = ao[src], vc = ac[src];
        const float* w0 = cw + tap * 64;
#pragma unroll 8
        for (int f = 0; f < 32; ++f) lf[f] += va * w0[f] + vc * w0[32 + f];
      }
    }
    const float* pvrow = pv + ((long long)b * T_ENC + ti) * ATTN_DIM;
    float s = v_b[0];
    for (int d = 0; d < ATTN_DIM; ++d) {
      float av = pvrow[d] + q[d];
      const float* dwd = dw + d;
#pragma unroll 8
      for (int f = 0; f < 32; ++f) av += lf[f] * dwd[f * ATTN_DIM];
      s += vw[d] * tanhf(av);
    }
    my_s[half] = s;
  }
  __syncthreads();   // all reads of a_old/a_cum complete

  // softmax over 512
  red[tid] = fmaxf(my_s[0], my_s[1]);
  __syncthreads();
  for (int off = 128; off > 0; off >>= 1) {
    if (tid < off) red[tid] = fmaxf(red[tid], red[tid + off]);
    __syncthreads();
  }
  float mx = red[0];
  __syncthreads();
  float e0 = expf(my_s[0] - mx), e1 = expf(my_s[1] - mx);
  red[tid] = e0 + e1;
  __syncthreads();
  for (int off = 128; off > 0; off >>= 1) {
    if (tid < off) red[tid] += red[tid + off];
    __syncthreads();
  }
  float inv = 1.0f / red[0];
  __syncthreads();
  float at0 = e0 * inv, at1 = e1 * inv;
  sc[tid] = at0; sc[tid + 256] = at1;

  long long aoff = ((long long)b * NUM_ITER + t) * T_ENC;
  attn_out[aoff + tid]       = at0;
  attn_out[aoff + tid + 256] = at1;
  a_cum[(long long)b * T_ENC + tid]       += at0;
  a_cum[(long long)b * T_ENC + tid + 256] += at1;
  a_old[(long long)b * T_ENC + tid]       = at0;
  a_old[(long long)b * T_ENC + tid + 256] = at1;
  __syncthreads();

  // context = attn @ memory
  for (int d = tid; d < D_ENC; d += 256) {
    float s = 0.0f;
    const float* mem = memory + (long long)b * T_ENC * D_ENC + d;
    for (int tt = 0; tt < T_ENC; ++tt) s += sc[tt] * mem[(long long)tt * D_ENC];
    context[(long long)b * D_ENC + d] = s;
  }
}

// ---------------------------------------------------------------------------
// Projection + stop token; writes frames & stop into d_out
// ---------------------------------------------------------------------------
__global__ __launch_bounds__(256)
void proj_k(const float* __restrict__ h2, const float* __restrict__ context,
            const float* __restrict__ proj_w, const float* __restrict__ proj_b,
            const float* __restrict__ stop_w, const float* __restrict__ stop_b,
            float* __restrict__ out_frames, float* __restrict__ out_stops, int t) {
  __shared__ float outv[R_FR];
  __shared__ float red[256];
  int b = blockIdx.x, tid = threadIdx.x;
  const float* h = h2 + (long long)b * DD_DIM;
  const float* cx = context + (long long)b * D_ENC;
  if (tid < R_FR) {
    float s = proj_b[tid];
    for (int k = 0; k < DD_DIM; ++k) s += h[k] * proj_w[(long long)k * R_FR + tid];
    for (int k = 0; k < D_ENC; ++k)  s += cx[k] * proj_w[(long long)(DD_DIM + k) * R_FR + tid];
    outv[tid] = s;
    out_frames[(long long)b * (T_DEC * FRAME) + (long long)t * R_FR + tid] = s;
  }
  __syncthreads();
  float s = 0.0f;
  for (int k = tid; k < DD_DIM; k += 256) s += h[k] * stop_w[k];
  for (int k = tid; k < R_FR; k += 256)   s += outv[k] * stop_w[DD_DIM + k];
  red[tid] = s;
  __syncthreads();
  for (int off = 128; off > 0; off >>= 1) {
    if (tid < off) red[tid] += red[tid + off];
    __syncthreads();
  }
  if (tid == 0) out_stops[(long long)b * NUM_ITER + t] = red[0] + stop_b[0];
}

// ---------------------------------------------------------------------------
// Host driver
// ---------------------------------------------------------------------------
extern "C" void kernel_launch(void* const* d_in, const int* in_sizes, int n_in,
                              void* d_out, int out_size, void* d_ws, size_t ws_size,
                              hipStream_t stream) {
  (void)in_sizes; (void)n_in; (void)out_size; (void)ws_size;
  const float* memory    = (const float*)d_in[0];
  const float* frames    = (const float*)d_in[1];
  const float* prenet_w1 = (const float*)d_in[2];
  const float* prenet_w2 = (const float*)d_in[3];
  const float* arnn_wx   = (const float*)d_in[4];
  const float* arnn_wh   = (const float*)d_in[5];
  const float* arnn_b    = (const float*)d_in[6];
  const float* attn_wq   = (const float*)d_in[7];
  const float* attn_wi   = (const float*)d_in[8];
  const float* attn_v_w  = (const float*)d_in[9];
  const float* attn_v_b  = (const float*)d_in[10];
  const float* loc_conv_w  = (const float*)d_in[11];
  const float* loc_dense_w = (const float*)d_in[12];
  const float* drnn_wx   = (const float*)d_in[13];
  const float* drnn_wh   = (const float*)d_in[14];
  const float* drnn_b    = (const float*)d_in[15];
  const float* proj_w    = (const float*)d_in[16];
  const float* proj_b    = (const float*)d_in[17];
  const float* stop_w    = (const float*)d_in[18];
  const float* stop_b    = (const float*)d_in[19];

  float* out_frames = (float*)d_out;                                   // [64,500,80]
  float* out_stops  = out_frames + (long long)B_SZ * T_DEC * FRAME;    // [64,250]
  float* out_attns  = out_stops + (long long)B_SZ * NUM_ITER;          // [64,250,512]

  // ---- workspace layout ----
  char* ws = (char*)d_ws;
  size_t off = 0;
  auto alloc = [&](size_t bytes) -> void* {
    void* p = ws + off;
    off = (off + bytes + 255) & ~(size_t)255;
    return p;
  };
  __bf16* W1p    = (__bf16*)alloc((size_t)K1 * N_GATE * 2);
  __bf16* W2p    = (__bf16*)alloc((size_t)K2 * N_GATE * 2);
  __bf16* Wqp    = (__bf16*)alloc((size_t)Q_DIM * ATTN_DIM * 2);
  __bf16* pre_bf = (__bf16*)alloc((size_t)B_SZ * NUM_ITER * PRE_DIM * 2);
  float*  pv     = (float*)alloc((size_t)B_SZ * T_ENC * ATTN_DIM * 4); // also prenet tmp
  __bf16* A1     = (__bf16*)alloc((size_t)B_SZ * K1 * 2);
  __bf16* A2     = (__bf16*)alloc((size_t)B_SZ * K2 * 2);
  float*  z      = (float*)alloc((size_t)B_SZ * N_GATE * 4);
  float*  procq  = (float*)alloc((size_t)B_SZ * ATTN_DIM * 4);
  float*  h1     = (float*)alloc((size_t)B_SZ * Q_DIM * 4);
  float*  c1     = (float*)alloc((size_t)B_SZ * Q_DIM * 4);
  float*  h2     = (float*)alloc((size_t)B_SZ * DD_DIM * 4);
  float*  c2     = (float*)alloc((size_t)B_SZ * DD_DIM * 4);
  __bf16* h1_bf  = (__bf16*)alloc((size_t)B_SZ * Q_DIM * 2);
  __bf16* h2_bf  = (__bf16*)alloc((size_t)B_SZ * DD_DIM * 2);
  float*  context= (float*)alloc((size_t)B_SZ * D_ENC * 4);
  float*  a_cum  = (float*)alloc((size_t)B_SZ * T_ENC * 4);
  float*  a_old  = (float*)alloc((size_t)B_SZ * T_ENC * 4);

  auto gridFor = [](long long n) { return (unsigned)((n + 255) / 256); };

  // ---- zero state ----
  zero_f32_k<<<gridFor(B_SZ * Q_DIM), 256, 0, stream>>>(h1, B_SZ * Q_DIM);
  zero_f32_k<<<gridFor(B_SZ * Q_DIM), 256, 0, stream>>>(c1, B_SZ * Q_DIM);
  zero_f32_k<<<gridFor(B_SZ * DD_DIM), 256, 0, stream>>>(h2, B_SZ * DD_DIM);
  zero_f32_k<<<gridFor(B_SZ * DD_DIM), 256, 0, stream>>>(c2, B_SZ * DD_DIM);
  zero_bf16_k<<<gridFor(B_SZ * Q_DIM), 256, 0, stream>>>(h1_bf, B_SZ * Q_DIM);
  zero_bf16_k<<<gridFor(B_SZ * DD_DIM), 256, 0, stream>>>(h2_bf, B_SZ * DD_DIM);
  zero_f32_k<<<gridFor(B_SZ * D_ENC), 256, 0, stream>>>(context, B_SZ * D_ENC);
  zero_f32_k<<<gridFor(B_SZ * T_ENC), 256, 0, stream>>>(a_cum, B_SZ * T_ENC);
  zero_f32_k<<<gridFor(B_SZ * T_ENC), 256, 0, stream>>>(a_old, B_SZ * T_ENC);

  // ---- one-time precompute ----
  pack_weights_k<<<gridFor((long long)K1 * N_GATE), 256, 0, stream>>>(
      arnn_wx, arnn_wh, 768, 1024, N_GATE, W1p);
  pack_weights_k<<<gridFor((long long)K2 * N_GATE), 256, 0, stream>>>(
      drnn_wx, drnn_wh, 1536, 1024, N_GATE, W2p);
  pack_weights_k<<<gridFor((long long)Q_DIM * ATTN_DIM), 256, 0, stream>>>(
      attn_wq, attn_wq, Q_DIM, 0, ATTN_DIM, Wqp);

  prenet1_k<<<gridFor((long long)B_SZ * NUM_ITER * PRE_DIM), 256, 0, stream>>>(
      frames, prenet_w1, pv /* tmp reuse */);
  prenet2_k<<<gridFor((long long)B_SZ * NUM_ITER * PRE_DIM), 256, 0, stream>>>(
      pv, prenet_w2, pre_bf);
  pv_k<<<gridFor((long long)B_SZ * T_ENC * ATTN_DIM), 256, 0, stream>>>(
      memory, attn_wi, pv);

  // ---- sequential decode loop ----
  const unsigned gGemmBig = (N_GATE / 16) / 8;              // 32 blocks (8 N-tiles each)
  const unsigned gGemmQ   = ((ATTN_DIM / 16) + 7) / 8;      // 1 block
  for (int t = 0; t < NUM_ITER; ++t) {
    assembleA1_k<<<gridFor((long long)B_SZ * K1), 256, 0, stream>>>(
        pre_bf, context, h1_bf, A1, t);
    gemm_wmma_bf16_k<<<gGemmBig, 256, 0, stream>>>(A1, W1p, arnn_b, z, K1, N_GATE);
    lstm_update_k<<<gridFor(B_SZ * Q_DIM), 256, 0, stream>>>(z, h1, c1, h1_bf);

    gemm_wmma_bf16_k<<<gGemmQ, 256, 0, stream>>>(h1_bf, Wqp, nullptr, procq,
                                                 Q_DIM, ATTN_DIM);
    attention_k<<<B_SZ, 256, 0, stream>>>(procq, pv, loc_conv_w, loc_dense_w,
                                          attn_v_w, attn_v_b, memory,
                                          a_cum, a_old, context, out_attns, t);

    assembleA2_k<<<gridFor((long long)B_SZ * K2), 256, 0, stream>>>(
        h1_bf, context, h2_bf, A2);
    gemm_wmma_bf16_k<<<gGemmBig, 256, 0, stream>>>(A2, W2p, drnn_b, z, K2, N_GATE);
    lstm_update_k<<<gridFor(B_SZ * DD_DIM), 256, 0, stream>>>(z, h2, c2, h2_bf);

    proj_k<<<B_SZ, 256, 0, stream>>>(h2, context, proj_w, proj_b,
                                     stop_w, stop_b, out_frames, out_stops, t);
  }
}
